// ConstantModulationAttention_45930380264095
// MI455X (gfx1250) — matrix operand order repaired and checked by reading, not compiled, not run on hardware
//
#include <hip/hip_runtime.h>
#include <hip/hip_bf16.h>

typedef __attribute__((ext_vector_type(16))) _Float16 v16h;
typedef __attribute__((ext_vector_type(8)))  _Float16 v8h;
typedef __attribute__((ext_vector_type(8)))  float    v8f;
typedef __attribute__((ext_vector_type(4)))  unsigned int u32x4;
typedef __attribute__((ext_vector_type(8)))  int          i32x8;
typedef __attribute__((ext_vector_type(4)))  int          i32x4;

#define B_  2
#define S_  2048
#define E_  1024
#define H_  16
#define D_  64
#define HD_ 1024
#define N_  (B_*S_)      // 4096 tokens
#define C3_ 3072         // 3*H*D qkv columns

#if defined(__HIP_DEVICE_COMPILE__) && __has_builtin(__builtin_amdgcn_tensor_load_to_lds)
#define USE_TDM 1
#else
#define USE_TDM 0
#endif

union V16U { v16h v; v8h h[2]; };

__device__ __forceinline__ v8f splat8(float x) {
  v8f r;
#pragma unroll
  for (int i = 0; i < 8; ++i) r[i] = x;
  return r;
}
__device__ __forceinline__ v8f exp8(v8f x) {
  v8f r;
#pragma unroll
  for (int i = 0; i < 8; ++i) r[i] = __expf(x[i]);
  return r;
}
__device__ __forceinline__ v8f max8(v8f a, v8f b) {
  v8f r;
#pragma unroll
  for (int i = 0; i < 8; ++i) r[i] = fmaxf(a[i], b[i]);
  return r;
}

__device__ __forceinline__ v8f wmma16(v16h a, v16h b, v8f c) {
  // D = A(16x32 f16) * B(32x16 f16) + C(16x16 f32)
  return __builtin_amdgcn_wmma_f32_16x16x32_f16(
      false, a, false, b, (short)0, c, false, false);
}

// A fragment: row-major LDS tile, 16 rows at mr, K window [kb, kb+32)
// lanes 0-15 hold K {kb..+7, kb+16..+23}; lanes 16-31 hold {+8, +24}
__device__ __forceinline__ v16h ldsA(const _Float16* base, int stride, int mr, int kb, int lane) {
  int r  = mr + (lane & 15);
  int kg = (lane & 16) ? 8 : 0;
  const _Float16* p = base + r * stride + kb + kg;
  V16U u;
  u.h[0] = *(const v8h*)(p);
  u.h[1] = *(const v8h*)(p + 16);
  return u.v;
}

// B fragment: LDS tile stored column-major BT[n][k]; 16 cols at nc, K window [kb, kb+32)
// lanes 0-15 hold K {kb..+7, kb+16..+23}; lanes 16-31 hold {+8, +24}
__device__ __forceinline__ v16h ldsB(const _Float16* base, int stride, int nc, int kb, int lane) {
  int c  = nc + (lane & 15);
  int ko = (lane & 16) ? 8 : 0;
  const _Float16* p = base + c * stride + kb + ko;
  V16U u;
  u.h[0] = *(const v8h*)(p);
  u.h[1] = *(const v8h*)(p + 16);
  return u.v;
}

#if USE_TDM
// TDM 2D f16 tile load: rowElems (x) * rows (y), global row stride rowStrideElems.
// LDS padding: +4 DWORDs after every 32 DWORDs (128B row) -> 72-half LDS stride.
// D# bit layout per CDNA5 ISA 8.3/8.4 (group0: count/lds/global/type, group1: dims).
// This toolchain's builtin takes 6 args: (g0, g1, g2, g3, extra, cpol).
__device__ __forceinline__ void tdm_load_2d(const _Float16* gsrc, unsigned lds_off,
                                            unsigned rows, unsigned rowElems,
                                            unsigned rowStrideElems)
{
  unsigned long long ga = (unsigned long long)(const void*)gsrc;
  u32x4 g0;
  g0[0] = 1u;                                              // count=1 (valid user D#)
  g0[1] = lds_off;                                         // LDS byte address
  g0[2] = (unsigned)(ga & 0xFFFFFFFFull);                  // global_addr[31:0]
  g0[3] = (unsigned)((ga >> 32) & 0x1FFFFFFull) | (2u << 30); // addr[56:32] | type=2
  i32x8 g1;
  g1[0] = (1 << 16)      // data_size = 1 -> 2 bytes
        | (1 << 20)      // pad_enable
        | (4 << 22)      // pad_interval: 32 DWORDs (128B) between pads
        | (3 << 25);     // pad_amount: 4 DWORDs (16B)
  g1[1] = (int)((rowElems & 0xFFFFu) << 16);               // tensor_dim0[15:0] @ [63:48]
  g1[2] = (int)(((rowElems >> 16) & 0xFFFFu) | ((rows & 0xFFFFu) << 16)); // dim0 hi | dim1 lo
  g1[3] = (int)(((rows >> 16) & 0xFFFFu) | ((rowElems & 0xFFFFu) << 16)); // dim1 hi | tile_dim0
  g1[4] = (int)(rows & 0xFFFFu);                           // tile_dim1 | tile_dim2=0
  g1[5] = (int)rowStrideElems;                             // tensor_dim0_stride[31:0]
  g1[6] = 0;                                               // stride hi | dim1_stride lo
  g1[7] = 0;
  i32x4 z4 = {};
  i32x8 z8 = {};
  __builtin_amdgcn_tensor_load_to_lds(g0, g1, z4, z4, z8, 0);
}
#endif

// ---------------------------------------------------------------------------
// Kernel 1: qkv = (x @ W_qkv) * sigmoid(scale_qkv); scatter to Q (pre-scaled by
// 1/sqrt(D)) / K as (b,h,s,d) and V transposed (b,h,d,s), all f16.
// 128x64 C-tile, 8 waves x (32x32), double-buffered LDS staging.
// ---------------------------------------------------------------------------
__global__ __launch_bounds__(256) void qkv_proj_kernel(
    const float* __restrict__ x, const float* __restrict__ Wqkv,
    const float* __restrict__ scale_qkv,
    _Float16* __restrict__ Qf, _Float16* __restrict__ Kf, _Float16* __restrict__ Vtf)
{
  __shared__ __align__(16) _Float16 As [2][128 * 40];
  __shared__ __align__(16) _Float16 BTs[2][64 * 40];
  const int tid = threadIdx.x, lane = tid & 31, wave = tid >> 5;
  const int cb = blockIdx.x * 64;           // column block in [0, 3072)
  const int rb = blockIdx.y * 128;          // token-row block in [0, 4096)
  const int mr = (wave & 3) * 32;
  const int nc = (wave >> 2) * 32;
  v8f acc[2][2] = {};
  const int NK = E_ / 32;

  auto stage = [&](int kb, int buf) {
#pragma unroll
    for (int i = 0; i < 16; ++i) {          // A: f32 -> f16, 128x32
      int e = tid + 256 * i; int r = e >> 5, kk = e & 31;
      As[buf][r * 40 + kk] = (_Float16)x[(size_t)(rb + r) * E_ + kb + kk];
    }
#pragma unroll
    for (int i = 0; i < 8; ++i) {           // B transposed: BT[n][k], 64x32
      int e = tid + 256 * i; int kk = e >> 6, nn = e & 63;
      BTs[buf][nn * 40 + kk] = (_Float16)Wqkv[(size_t)(kb + kk) * C3_ + cb + nn];
    }
  };

  stage(0, 0);
  for (int it = 0; it < NK; ++it) {
    __syncthreads();
    if (it + 2 < NK) {                      // prefetch 2 K-steps ahead
      __builtin_prefetch(&x[(size_t)(rb + (tid & 127)) * E_ + (it + 2) * 32], 0, 1);
      __builtin_prefetch(&Wqkv[(size_t)((it + 2) * 32 + (lane & 31)) * C3_ + cb], 0, 1);
    }
    if (it + 1 < NK) stage((it + 1) * 32, (it + 1) & 1);
    const _Float16* Ab = As [it & 1];
    const _Float16* Bb = BTs[it & 1];
    v16h a0 = ldsA(Ab, 40, mr,      0, lane);
    v16h a1 = ldsA(Ab, 40, mr + 16, 0, lane);
    v16h b0 = ldsB(Bb, 40, nc,      0, lane);
    v16h b1 = ldsB(Bb, 40, nc + 16, 0, lane);
    acc[0][0] = wmma16(a0, b0, acc[0][0]);
    acc[0][1] = wmma16(a0, b1, acc[0][1]);
    acc[1][0] = wmma16(a1, b0, acc[1][0]);
    acc[1][1] = wmma16(a1, b1, acc[1][1]);
  }

  const int hi = (lane & 16) ? 8 : 0;
#pragma unroll
  for (int rf = 0; rf < 2; ++rf) {
#pragma unroll
    for (int f = 0; f < 2; ++f) {
      const v8f a = acc[rf][f];
      int c  = cb + nc + f * 16 + (lane & 15);
      int q  = c >> 10, hd = c & 1023;
      int h  = hd >> 6, d = hd & 63;
      float sc = 1.0f / (1.0f + __expf(-scale_qkv[hd]));
      if (q == 0) sc *= 0.125f;             // fold 1/sqrt(D) into Q
#pragma unroll
      for (int i = 0; i < 8; ++i) {
        int n = rb + mr + rf * 16 + hi + i;
        int b = n >> 11, s = n & 2047;
        _Float16 v = (_Float16)(a[i] * sc);
        size_t bh = (size_t)(b * H_ + h);
        if (q == 0)      Qf [bh * (S_ * D_) + (size_t)s * D_ + d] = v;
        else if (q == 1) Kf [bh * (S_ * D_) + (size_t)s * D_ + d] = v;
        else             Vtf[bh * (D_ * S_) + (size_t)d * S_ + s] = v;
      }
    }
  }
}

// ---------------------------------------------------------------------------
// Kernel 2: causal flash attention per (b, h, 128-row s-block).
// Q/K/V^T tiles staged by the Tensor Data Mover when available.
// ---------------------------------------------------------------------------
__global__ __launch_bounds__(256) void flash_attn_kernel(
    const _Float16* __restrict__ Qg, const _Float16* __restrict__ Kg,
    const _Float16* __restrict__ Vtg, _Float16* __restrict__ attn)
{
  __shared__ __align__(16) _Float16 Qs [128 * 72];    // q rows x d (pre-scaled)
  __shared__ __align__(16) _Float16 Ks [64 * 72];     // t rows x d  (BT for Q.K^T)
  __shared__ __align__(16) _Float16 Vts[64 * 72];     // d rows x t  (BT for P.V)
  __shared__ __align__(16) _Float16 Ps [8 * 16 * 72]; // per-wave P scratch
  const int tid = threadIdx.x, lane = tid & 31, wave = tid >> 5;
  const int sb = blockIdx.x, h = blockIdx.y, b = blockIdx.z;
  const int s0 = sb * 128;
  const size_t bh = (size_t)(b * H_ + h);
  const _Float16* Qbh = Qg  + bh * (S_ * D_);
  const _Float16* Kbh = Kg  + bh * (S_ * D_);
  const _Float16* Vbh = Vtg + bh * (D_ * S_);

#if USE_TDM
  if (wave == 0) {
    tdm_load_2d(Qbh + (size_t)s0 * D_, (unsigned)(unsigned long long)(void*)Qs,
                /*rows=*/128, /*rowElems=*/64, /*rowStride=*/64);
    __builtin_amdgcn_s_wait_tensorcnt(0);
  }
#else
#pragma unroll
  for (int i = 0; i < 32; ++i) {
    int e = tid + 256 * i; int r = e >> 6, d = e & 63;
    Qs[r * 72 + d] = Qbh[(size_t)(s0 + r) * D_ + d];
  }
#endif
  __syncthreads();

  const int mr = wave * 16;
  const int hi = (lane & 16) ? 8 : 0;
  v16h qa0 = ldsA(Qs, 72, mr, 0,  lane);   // Q tile immutable: hoist A-frags
  v16h qa1 = ldsA(Qs, 72, mr, 32, lane);
  v8f m = splat8(-3.0e38f);
  v8f l = {};
  v8f acc[4] = {};

  const int nT = sb * 2 + 2;               // causal: only t <= s0+127
  for (int tb = 0; tb < nT; ++tb) {
    const int t0 = tb * 64;
    __syncthreads();                       // prior iteration's reads done
#if USE_TDM
    if (wave == 0) {
      tdm_load_2d(Kbh + (size_t)t0 * D_, (unsigned)(unsigned long long)(void*)Ks,
                  64, 64, 64);
      tdm_load_2d(Vbh + t0, (unsigned)(unsigned long long)(void*)Vts,
                  64, 64, S_);
      __builtin_amdgcn_s_wait_tensorcnt(0);
    }
#else
#pragma unroll
    for (int i = 0; i < 16; ++i) {         // K tile [t][d]
      int e = tid + 256 * i; int r = e >> 6, d = e & 63;
      Ks[r * 72 + d] = Kbh[(size_t)(t0 + r) * D_ + d];
    }
#pragma unroll
    for (int i = 0; i < 16; ++i) {         // V^T tile [d][t]
      int e = tid + 256 * i; int dd = e >> 6, t = e & 63;
      Vts[dd * 72 + t] = Vbh[(size_t)dd * S_ + t0 + t];
    }
#endif
    __syncthreads();

    v8f sf[4];
#pragma unroll
    for (int j = 0; j < 4; ++j) {          // S = Q.K^T  (M=s16, N=t64, K=d64)
      v8f c = {};
      c = wmma16(qa0, ldsB(Ks, 72, j * 16, 0,  lane), c);
      c = wmma16(qa1, ldsB(Ks, 72, j * 16, 32, lane), c);
      sf[j] = c;
    }

    const int srb = s0 + mr + hi;
    const int tcb = t0 + (lane & 15);
#pragma unroll
    for (int j = 0; j < 4; ++j) {          // causal mask
      int tc = tcb + j * 16;
#pragma unroll
      for (int i = 0; i < 8; ++i)
        if (tc > srb + i) sf[j][i] = -3.0e38f;
    }

    v8f rmax = max8(max8(sf[0], sf[1]), max8(sf[2], sf[3]));
#pragma unroll
    for (int i = 0; i < 8; ++i) {          // reduce within 16-lane half (same rows)
      float v = rmax[i];
      v = fmaxf(v, __shfl_xor(v, 1, 32));
      v = fmaxf(v, __shfl_xor(v, 2, 32));
      v = fmaxf(v, __shfl_xor(v, 4, 32));
      v = fmaxf(v, __shfl_xor(v, 8, 32));
      rmax[i] = v;
    }
    v8f mnew  = max8(m, rmax);
    v8f alpha = exp8(m - mnew);
    v8f p[4];
#pragma unroll
    for (int j = 0; j < 4; ++j) p[j] = exp8(sf[j] - mnew);
    v8f rsum = p[0] + p[1] + p[2] + p[3];
#pragma unroll
    for (int i = 0; i < 8; ++i) {
      float v = rsum[i];
      v += __shfl_xor(v, 1, 32);
      v += __shfl_xor(v, 2, 32);
      v += __shfl_xor(v, 4, 32);
      v += __shfl_xor(v, 8, 32);
      rsum[i] = v;
    }
    l = l * alpha + rsum;
    m = mnew;
#pragma unroll
    for (int j = 0; j < 4; ++j) acc[j] = acc[j] * alpha;

    // stage P (f16) into this wave's private LDS region, re-read as A-fragments
    _Float16* Pw = Ps + wave * (16 * 72);
#pragma unroll
    for (int j = 0; j < 4; ++j)
#pragma unroll
      for (int i = 0; i < 8; ++i)
        Pw[(hi + i) * 72 + j * 16 + (lane & 15)] = (_Float16)p[j][i];
    asm volatile("s_wait_dscnt 0" ::: "memory");  // same-wave LDS RAW fence

    v16h pa0 = ldsA(Pw, 72, 0, 0,  lane);
    v16h pa1 = ldsA(Pw, 72, 0, 32, lane);
#pragma unroll
    for (int j = 0; j < 4; ++j) {          // O += P.V  (M=s16, N=d64, K=t64)
      acc[j] = wmma16(pa0, ldsB(Vts, 72, j * 16, 0,  lane), acc[j]);
      acc[j] = wmma16(pa1, ldsB(Vts, 72, j * 16, 32, lane), acc[j]);
    }
  }

  // normalize and store attn as f16 [token][h*64+d] for the output GEMM
#pragma unroll
  for (int j = 0; j < 4; ++j) {
#pragma unroll
    for (int i = 0; i < 8; ++i) {
      int srow = s0 + mr + hi + i;
      size_t n = (size_t)b * S_ + srow;
      int col  = h * D_ + j * 16 + (lane & 15);
      attn[n * HD_ + col] = (_Float16)(acc[j][i] / l[i]);
    }
  }
}

// ---------------------------------------------------------------------------
// Kernel 3: out = (attn @ W_out) * sigmoid(scale_out), f32 output.
// Same 128x64 double-buffered WMMA tiling as kernel 1; A is already f16.
// ---------------------------------------------------------------------------
__global__ __launch_bounds__(256) void out_proj_kernel(
    const _Float16* __restrict__ attn, const float* __restrict__ Wout,
    const float* __restrict__ scale_out, float* __restrict__ out)
{
  __shared__ __align__(16) _Float16 As [2][128 * 40];
  __shared__ __align__(16) _Float16 BTs[2][64 * 40];
  const int tid = threadIdx.x, lane = tid & 31, wave = tid >> 5;
  const int cb = blockIdx.x * 64;          // output-feature block
  const int rb = blockIdx.y * 128;         // token block
  const int mr = (wave & 3) * 32;
  const int nc = (wave >> 2) * 32;
  v8f acc[2][2] = {};
  const int NK = HD_ / 32;

  auto stage = [&](int kb, int buf) {
#pragma unroll
    for (int i = 0; i < 16; ++i) {
      int e = tid + 256 * i; int r = e >> 5, kk = e & 31;
      As[buf][r * 40 + kk] = attn[(size_t)(rb + r) * HD_ + kb + kk];
    }
#pragma unroll
    for (int i = 0; i < 8; ++i) {
      int e = tid + 256 * i; int kk = e >> 6, nn = e & 63;
      BTs[buf][nn * 40 + kk] = (_Float16)Wout[(size_t)(kb + kk) * E_ + cb + nn];
    }
  };

  stage(0, 0);
  for (int it = 0; it < NK; ++it) {
    __syncthreads();
    if (it + 2 < NK) {
      __builtin_prefetch(&attn[(size_t)(rb + (tid & 127)) * HD_ + (it + 2) * 32], 0, 1);
      __builtin_prefetch(&Wout[(size_t)((it + 2) * 32 + (lane & 31)) * E_ + cb], 0, 1);
    }
    if (it + 1 < NK) stage((it + 1) * 32, (it + 1) & 1);
    const _Float16* Ab = As [it & 1];
    const _Float16* Bb = BTs[it & 1];
    v16h a0 = ldsA(Ab, 40, mr,      0, lane);
    v16h a1 = ldsA(Ab, 40, mr + 16, 0, lane);
    v16h b0 = ldsB(Bb, 40, nc,      0, lane);
    v16h b1 = ldsB(Bb, 40, nc + 16, 0, lane);
    acc[0][0] = wmma16(a0, b0, acc[0][0]);
    acc[0][1] = wmma16(a0, b1, acc[0][1]);
    acc[1][0] = wmma16(a1, b0, acc[1][0]);
    acc[1][1] = wmma16(a1, b1, acc[1][1]);
  }

  const int hi = (lane & 16) ? 8 : 0;
#pragma unroll
  for (int rf = 0; rf < 2; ++rf) {
#pragma unroll
    for (int f = 0; f < 2; ++f) {
      const v8f a = acc[rf][f];
      int c = cb + nc + f * 16 + (lane & 15);
      float so = 1.0f / (1.0f + __expf(-scale_out[c]));
#pragma unroll
      for (int i = 0; i < 8; ++i) {
        int n = rb + mr + rf * 16 + hi + i;
        out[(size_t)n * E_ + c] = a[i] * so;
      }
    }
  }
}

// ---------------------------------------------------------------------------
extern "C" void kernel_launch(void* const* d_in, const int* in_sizes, int n_in,
                              void* d_out, int out_size, void* d_ws, size_t ws_size,
                              hipStream_t stream) {
  const float* x         = (const float*)d_in[0];
  const float* W_qkv     = (const float*)d_in[1];
  const float* W_out     = (const float*)d_in[2];
  const float* scale_qkv = (const float*)d_in[3];
  const float* scale_out = (const float*)d_in[4];
  // d_in[5] = mask: causal, computed analytically in-kernel.

  const size_t perTensor = (size_t)B_ * H_ * S_ * D_;   // 4,194,304 halves (8 MB)
  _Float16* Qf   = (_Float16*)d_ws;
  _Float16* Kf   = Qf  + perTensor;
  _Float16* Vtf  = Kf  + perTensor;
  _Float16* attn = Vtf + perTensor;                     // total ws: 32 MB

  qkv_proj_kernel  <<<dim3(C3_ / 64, N_ / 128), 256, 0, stream>>>(x, W_qkv, scale_qkv, Qf, Kf, Vtf);
  flash_attn_kernel<<<dim3(S_ / 128, H_, B_),   256, 0, stream>>>(Qf, Kf, Vtf, attn);
  out_proj_kernel  <<<dim3(E_ / 64, N_ / 128),  256, 0, stream>>>(attn, W_out, scale_out, (float*)d_out);
}
